// Attention_36412732735651
// MI455X (gfx1250) — matrix operand order repaired and checked by reading, not compiled, not run on hardware
//
#include <hip/hip_runtime.h>
#include <hip/hip_bf16.h>
#include <math.h>

typedef __attribute__((ext_vector_type(16))) _Float16 v16h;
typedef __attribute__((ext_vector_type(4)))  _Float16 v4h;
typedef __attribute__((ext_vector_type(8)))  float    v8f;
typedef __attribute__((ext_vector_type(4)))  float    v4f;

#define ATT_B 4
#define ATT_H 16
#define ATT_S 2048
#define ATT_D 64

// LDS pitches (padded to avoid bank conflicts on the f32 score panel)
#define SC_PITCH 2052   // floats (multiple of 4 -> 16B-aligned float4 rows)
#define PF_PITCH 2056   // halfs  (multiple of 8 -> aligned vector reads)

// LDS layout offsets (bytes)
#define SC_BYTES   (16 * SC_PITCH * 4)          // 131328
#define PF_BYTES   (16 * PF_PITCH * 2)          // 65792
#define RED_BYTES  (16 * 16 * 4)                // 1024
#define RMAX_BYTES (16 * 4)
#define RINV_BYTES (16 * 4)
#define OACC_BYTES (16 * 64 * 4)                // 4096
#define SMEM_TOTAL (SC_BYTES + PF_BYTES + RED_BYTES + RMAX_BYTES + RINV_BYTES + OACC_BYTES)

__global__ __launch_bounds__(256)
void Attention_36412732735651_kernel(const float* __restrict__ q,
                                     const float* __restrict__ k,
                                     const float* __restrict__ v,
                                     const int*   __restrict__ mask,
                                     float* __restrict__ out,
                                     float* __restrict__ p_attn)
{
    extern __shared__ char raw[];
    float*    sc     = (float*)raw;                                   // [16][SC_PITCH] f32 scores
    _Float16* pf     = (_Float16*)(raw + SC_BYTES);                   // [16][PF_PITCH] f16 P
    float*    red    = (float*)(raw + SC_BYTES + PF_BYTES);           // [16][16] partials
    float*    rowmax = (float*)(raw + SC_BYTES + PF_BYTES + RED_BYTES);
    float*    rowinv = (float*)(raw + SC_BYTES + PF_BYTES + RED_BYTES + RMAX_BYTES);
    float*    oacc   = (float*)(raw + SC_BYTES + PF_BYTES + RED_BYTES + RMAX_BYTES + RINV_BYTES); // [16][64]

    const int t    = threadIdx.x;          // 0..255 (8 wave32s)
    const int lane = t & 31;
    const int wv   = t >> 5;

    const int tiles_per_bh = ATT_S / 16;   // 128
    const int bh  = blockIdx.x / tiles_per_bh;
    const int m0  = (blockIdx.x % tiles_per_bh) * 16;
    const int b   = bh / ATT_H;

    const size_t bh_off = (size_t)bh * ATT_S * ATT_D;
    const float* Qb = q + bh_off;
    const float* Kb = k + bh_off;
    const float* Vb = v + bh_off;
    const int*   Mb = mask + (size_t)b * ATT_S;

    // --- WMMA lane mappings (ISA 7.12.2, wave32) ---
    const int frow = lane & 15;            // A row / B,C,D column
    const int koff = (lane >> 4) * 8;      // A-matrix K sub-offset
    const int krow = (lane >> 4) * 16;     // unused; B uses koff, same k order as A
    const float scale = 0.125f;            // 1/sqrt(64)

    // ---------------- Phase 1: scores = Q K^T (f16 WMMA, f32 accum) ----------------
    v16h aq0, aq1;
    {
        const float* Qrow = Qb + (size_t)(m0 + frow) * ATT_D;
        #pragma unroll
        for (int j = 0; j < 16; ++j) {
            const int dj = (j < 8) ? (koff + j) : (16 + koff + (j - 8));
            aq0[j] = (_Float16)Qrow[dj];
            aq1[j] = (_Float16)Qrow[32 + dj];
        }
    }

    // 8 waves stripe the key dimension: 16 iterations x (8 waves * 16 keys) = 2048
    for (int it = 0; it < ATT_S / 128; ++it) {
        const int n0 = (it * 8 + wv) * 16;
        const float* Krow = Kb + (size_t)(n0 + frow) * ATT_D;
        if (it + 1 < ATT_S / 128)
            __builtin_prefetch(Krow + 128 * ATT_D, 0, 0);   // next K tile -> global_prefetch_b8

        v16h bk0, bk1;
        #pragma unroll
        for (int j = 0; j < 16; ++j) {
            const int dj = (j < 8) ? (koff + j) : (16 + koff + (j - 8));
            bk0[j] = (_Float16)Krow[dj];
            bk1[j] = (_Float16)Krow[32 + dj];
        }

        v8f c = {};
        c = __builtin_amdgcn_wmma_f32_16x16x32_f16(false, aq0, false, bk0, (short)0, c, false, false);
        c = __builtin_amdgcn_wmma_f32_16x16x32_f16(false, aq1, false, bk1, (short)0, c, false, false);

        const int mval = Mb[n0 + frow];    // mask depends only on key index
        #pragma unroll
        for (int i = 0; i < 8; ++i) {
            const int M = (lane >> 4) * 8 + i;
            float s = c[i] * scale;
            if (mval == 0) s = -1.0e9f;
            sc[M * SC_PITCH + n0 + frow] = s;
        }
    }
    __syncthreads();

    // ---------------- Phase 2: softmax over S (fp32, in LDS, float4-wide) ----------------
    const int r   = t >> 4;                // query row 0..15
    const int seg = t & 15;                // 16 threads per row

    float mx = -3.4e38f;
    for (int c2 = seg * 4; c2 < ATT_S; c2 += 64) {
        const v4f s4 = *(const v4f*)&sc[r * SC_PITCH + c2];
        mx = fmaxf(mx, fmaxf(fmaxf(s4.x, s4.y), fmaxf(s4.z, s4.w)));
    }
    red[r * 16 + seg] = mx;
    __syncthreads();
    if (t < 16) {
        float m2 = red[t * 16];
        #pragma unroll
        for (int j = 1; j < 16; ++j) m2 = fmaxf(m2, red[t * 16 + j]);
        rowmax[t] = m2;
    }
    __syncthreads();

    const float rm = rowmax[r];
    float sum = 0.0f;
    for (int c2 = seg * 4; c2 < ATT_S; c2 += 64) {
        v4f e4 = *(const v4f*)&sc[r * SC_PITCH + c2];
        e4.x = __expf(e4.x - rm);
        e4.y = __expf(e4.y - rm);
        e4.z = __expf(e4.z - rm);
        e4.w = __expf(e4.w - rm);
        *(v4f*)&sc[r * SC_PITCH + c2] = e4;
        sum += e4.x + e4.y + e4.z + e4.w;
    }
    red[r * 16 + seg] = sum;
    __syncthreads();
    if (t < 16) {
        float s2 = red[t * 16];
        #pragma unroll
        for (int j = 1; j < 16; ++j) s2 += red[t * 16 + j];
        rowinv[t] = 1.0f / s2;
    }
    __syncthreads();

    // Normalize: ONE streaming pass -> 128b non-temporal HBM stores of p_attn
    // (write-once data; NT hint keeps Q/K/V resident in the 192MB L2),
    // plus an f16 copy into LDS for the P*V GEMM.
    for (int r2 = 0; r2 < 16; ++r2) {
        const float ri = rowinv[r2];
        float* prow = p_attn + ((size_t)bh * ATT_S + m0 + r2) * ATT_S;
        #pragma unroll
        for (int c2 = t * 4; c2 < ATT_S; c2 += 1024) {
            const v4f e4 = *(const v4f*)&sc[r2 * SC_PITCH + c2];
            v4f p4;
            p4.x = e4.x * ri; p4.y = e4.y * ri; p4.z = e4.z * ri; p4.w = e4.w * ri;
            __builtin_nontemporal_store(p4, (v4f*)&prow[c2]);   // global_store_b128, NT
            v4h h4;
            h4.x = (_Float16)p4.x; h4.y = (_Float16)p4.y;
            h4.z = (_Float16)p4.z; h4.w = (_Float16)p4.w;
            *(v4h*)&pf[r2 * PF_PITCH + c2] = h4;                // 8B ds store
        }
    }
    {   // zero the split-K accumulator (one float4 per thread)
        v4f z = {0.0f, 0.0f, 0.0f, 0.0f};
        *(v4f*)&oacc[t * 4] = z;
    }
    __syncthreads();

    // ---------------- Phase 3: out = P V (f16 WMMA from LDS, split-K) ----------------
    // 8 waves: 4 output N-tiles (16 of D=64) x 2 K-halves of 1024; ds_add_f32 reduction.
    const int ntile = wv & 3;
    const int khalf = wv >> 2;
    const int n0v   = ntile * 16;

    v8f co = {};
    const int kk_beg = khalf * (ATT_S / 2);
    const int kk_end = kk_beg + (ATT_S / 2);
    for (int kk = kk_beg; kk < kk_end; kk += 32) {
        v16h ap, bv;
        #pragma unroll
        for (int j = 0; j < 16; ++j) {
            const int kj = (j < 8) ? (kk + koff + j) : (kk + 16 + koff + (j - 8));
            ap[j] = pf[frow * PF_PITCH + kj];
        }
        #pragma unroll
        for (int j = 0; j < 16; ++j)
            bv[j] = (_Float16)Vb[(size_t)((j < 8) ? (kk + koff + j) : (kk + 16 + koff + (j - 8))) * ATT_D + n0v + frow];

        co = __builtin_amdgcn_wmma_f32_16x16x32_f16(false, ap, false, bv, (short)0, co, false, false);
    }
    #pragma unroll
    for (int i = 0; i < 8; ++i) {
        const int M = (lane >> 4) * 8 + i;
        atomicAdd(&oacc[M * 64 + n0v + frow], co[i]);   // ds_add_f32
    }
    __syncthreads();

    // out tile write: one 128b store per thread, fully coalesced
    float* ob = out + ((size_t)bh * ATT_S + m0) * ATT_D;
    *(v4f*)&ob[t * 4] = *(const v4f*)&oacc[t * 4];
}

extern "C" void kernel_launch(void* const* d_in, const int* in_sizes, int n_in,
                              void* d_out, int out_size, void* d_ws, size_t ws_size,
                              hipStream_t stream) {
    const float* q    = (const float*)d_in[0];
    const float* k    = (const float*)d_in[1];
    const float* v    = (const float*)d_in[2];
    const int*   mask = (const int*)d_in[3];

    float* out    = (float*)d_out;                                    // [B,H,S,D]
    float* p_attn = out + (size_t)ATT_B * ATT_H * ATT_S * ATT_D;      // [B,H,S,S]

    const int blocks = ATT_B * ATT_H * (ATT_S / 16);                  // 8192
    hipLaunchKernelGGL(Attention_36412732735651_kernel,
                       dim3(blocks), dim3(256), SMEM_TOTAL, stream,
                       q, k, v, mask, out, p_attn);
}
